// DCRNNModel_35837207118645
// MI455X (gfx1250) — matrix operand order repaired and checked by reading, not compiled, not run on hardware
//
#include <hip/hip_runtime.h>
#include <hip/hip_bf16.h>
#include <math.h>

// ---------------------------------------------------------------------------
// DCRNN (K=2 diffusion, HID=64, N=512, B=16, T_in=12, T_out=12) on gfx1250.
// All GEMMs run as bf16 WMMA (v_wmma_f32_16x16x32_bf16) with fp32 accumulate.
// GEMM form: C(M,N) = A(MxK) * Bt(NxK)^T, both operands K-major => all
// fragment loads are contiguous 16B (b128).  The wave tile is 64 x (16*NT)
// with NT a compile-time constant so all fragments live in VGPRs.
// ---------------------------------------------------------------------------

#define NND 512     // nodes
#define BB  16      // batch
#define HID 64
#define T_IN 12
#define T_OUT 12

typedef __attribute__((ext_vector_type(16))) __bf16 v16bf;
typedef __attribute__((ext_vector_type(8)))  float  v8f;

union FragAB { uint4 q[2]; v16bf v; };

__device__ __forceinline__ unsigned short f2bf(float f) {
  unsigned int u = __float_as_uint(f);
  u = (u + 0x7FFFu + ((u >> 16) & 1u)) >> 16;   // round-to-nearest-even
  return (unsigned short)u;
}

// ---------------------------------------------------------------------------
// WMMA GEMM: one wave (32 threads) per 64 x (16*NT) C tile.
//   A  : bf16 (M x K) row-major, lda = K
//   Bt : bf16 (N x K) row-major, ldb = K   (i.e. B stored transposed)
// Epilogues (any subset):
//   Cf   : fp32 out (M x ldc), + bias[n], act 0=none 1=sigmoid 2=tanh
//   OutXT: bf16 out at [col*xt_pitch + m]       (transposed state, next hop in)
//   OutG : bf16 out at [(m*16 + col/Freal)*g_pitch + g_coloff + col%Freal]
//          (scatter into concatenated gate-GEMM input)
// ---------------------------------------------------------------------------
template <int NT>
__global__ __launch_bounds__(32) void gemm_bf16_wmma(
    const unsigned short* __restrict__ A, int lda,
    const unsigned short* __restrict__ Bt, int ldb,
    int nbase0, int K,
    float* __restrict__ Cf, int ldc, const float* __restrict__ bias, int act,
    unsigned short* __restrict__ OutXT, int xt_pitch,
    unsigned short* __restrict__ OutG, int g_pitch, int g_coloff, int Freal)
{
  const int lane = threadIdx.x;
  const int sel  = lane >> 4;      // half-wave select
  const int l16  = lane & 15;
  const int mbase = blockIdx.x * 64;
  const int nbase = nbase0 + blockIdx.y * (16 * NT);

  v8f c[4][NT];
#pragma unroll
  for (int i = 0; i < 4; ++i)
#pragma unroll
    for (int j = 0; j < NT; ++j)
#pragma unroll
      for (int e = 0; e < 8; ++e) c[i][j][e] = 0.0f;

  const unsigned short* Arow = A + (size_t)(mbase + l16) * lda + sel * 8;
  const unsigned short* Brow = Bt + (size_t)(nbase + l16) * ldb + sel * 8;

  for (int k0 = 0; k0 < K; k0 += 32) {
    FragAB a[4], b[NT];
    // A fragment (16-bit 16x32 layout): half-wave 0 -> K {0..7,16..23},
    // half-wave 1 -> K {8..15,24..31}; each chunk one b128.
#pragma unroll
    for (int i = 0; i < 4; ++i) {
      const unsigned short* p = Arow + (size_t)i * 16 * lda + k0;
      a[i].q[0] = *(const uint4*)p;
      a[i].q[1] = *(const uint4*)(p + 16);
    }
    // B fragment: same K layout as A (half-wave h -> K {8h..8h+7, 16+8h..16+8h+7}).
#pragma unroll
    for (int j = 0; j < NT; ++j) {
      const unsigned short* p = Brow + (size_t)j * 16 * ldb + k0;
      b[j].q[0] = *(const uint4*)p;
      b[j].q[1] = *(const uint4*)(p + 16);
    }
    __builtin_prefetch((const void*)(Arow + k0 + 32), 0, 1);
    __builtin_prefetch((const void*)(Brow + k0 + 32), 0, 1);
#pragma unroll
    for (int i = 0; i < 4; ++i)
#pragma unroll
      for (int j = 0; j < NT; ++j)
        c[i][j] = __builtin_amdgcn_wmma_f32_16x16x32_bf16(
            false, a[i].v, false, b[j].v, (short)0, c[i][j], false, false);
  }

  // Epilogue: C element (m = mbase+i*16+e+8*sel, n = nbase+j*16+l16)
#pragma unroll
  for (int i = 0; i < 4; ++i) {
#pragma unroll
    for (int j = 0; j < NT; ++j) {
      const int col = nbase + j * 16 + l16;
#pragma unroll
      for (int e = 0; e < 8; ++e) {
        const int m = mbase + i * 16 + e + 8 * sel;
        const float v = c[i][j][e];
        if (Cf) {
          float x = v + (bias ? bias[col] : 0.0f);
          if (act == 1)      x = 1.0f / (1.0f + __expf(-x));
          else if (act == 2) x = tanhf(x);
          Cf[(size_t)m * ldc + col] = x;
        }
        if (OutXT || OutG) {
          const unsigned short bf = f2bf(v);
          if (OutXT) OutXT[(size_t)col * xt_pitch + m] = bf;
          if (OutG) {
            const int bb = col / Freal;
            const int ff = col - bb * Freal;
            OutG[(size_t)(m * BB + bb) * g_pitch + g_coloff + ff] = bf;
          }
        }
      }
    }
  }
}

// ---------------------------------------------------------------------------
// Build concatenated cell input:  slice0 of diffusion (X0T, transposed) and
// columns [0,F) of G;  also zero-fills G pad columns [5F, 5F+padcols).
//   f <  FA : from Aptr  (FA==1: xl = Aptr[b*Astride + j];
//                         else  : state Aptr[(j*16+b)*64 + f])
//   f >= FA : from hB[(j*16+b)*64 + f-FA]  (optionally * rmul[...])
// ---------------------------------------------------------------------------
__global__ void concat_kernel(const float* __restrict__ Aptr, int Astride, int FA,
                              const float* __restrict__ hB,
                              const float* __restrict__ rmul,
                              int F, int padcols, int gp,
                              unsigned short* __restrict__ X0T,
                              unsigned short* __restrict__ G)
{
  const int Ftot  = F + padcols;
  const int total = NND * BB * Ftot;
  for (int idx = blockIdx.x * blockDim.x + threadIdx.x; idx < total;
       idx += gridDim.x * blockDim.x) {
    const int f   = idx % Ftot;
    const int rem = idx / Ftot;
    const int b   = rem & (BB - 1);
    const int j   = rem >> 4;
    if (f >= F) {                       // zero G pad columns (layer-0 only)
      G[(size_t)(j * BB + b) * gp + 5 * F + (f - F)] = 0;
      continue;
    }
    float v;
    if (f < FA) {
      v = (FA == 1) ? Aptr[b * Astride + j]
                    : Aptr[((size_t)j * BB + b) * HID + f];
    } else {
      const size_t hi = ((size_t)j * BB + b) * HID + (f - FA);
      v = hB[hi];
      if (rmul) v *= rmul[hi];
    }
    const unsigned short bf = f2bf(v);
    X0T[(size_t)(b * F + f) * NND + j] = bf;        // transposed state
    G[(size_t)(j * BB + b) * gp + f]   = bf;        // gate-GEMM row
  }
}

// GRU state update: h = (1-z)*h + z*n   (fp32 master state)
__global__ void gru_update(float* __restrict__ h, const float* __restrict__ z,
                           const float* __restrict__ n, int total)
{
  for (int i = blockIdx.x * blockDim.x + threadIdx.x; i < total;
       i += gridDim.x * blockDim.x) {
    const float zz = z[i];
    h[i] = (1.0f - zz) * h[i] + zz * n[i];
  }
}

// Projection head: yt[b,j] = h1[j,b,:] . Wp + bp ; writes d_out slice + di
__global__ void head_kernel(const float* __restrict__ h1,
                            const float* __restrict__ Wp,
                            const float* __restrict__ bp,
                            float* __restrict__ out_t,   // d_out + t*N
                            float* __restrict__ di)
{
  const int idx = blockIdx.x * blockDim.x + threadIdx.x;
  if (idx >= BB * NND) return;
  const int b = idx / NND, j = idx % NND;
  float acc = bp[0];
  const float* hp = h1 + ((size_t)j * BB + b) * HID;
#pragma unroll 8
  for (int c0 = 0; c0 < HID; ++c0) acc += hp[c0] * Wp[c0];
  out_t[(size_t)b * (T_OUT * NND) + j] = acc;
  di[idx] = acc;
}

// Row sums of A and of A^T (one block per node, LDS reduction)
__global__ void rowcol_sums(const float* __restrict__ A,
                            float* __restrict__ rs, float* __restrict__ cs)
{
  __shared__ float sr[256], sc[256];
  const int i = blockIdx.x, t = threadIdx.x;
  float r = 0.0f, c = 0.0f;
  for (int j = t; j < NND; j += 256) {
    r += A[(size_t)i * NND + j];
    c += A[(size_t)j * NND + i];
  }
  sr[t] = r; sc[t] = c; __syncthreads();
  for (int s = 128; s > 0; s >>= 1) {
    if (t < s) { sr[t] += sr[t + s]; sc[t] += sc[t + s]; }
    __syncthreads();
  }
  if (t == 0) { rs[i] = sr[0] + 1e-6f; cs[i] = sc[0] + 1e-6f; }
}

// Wf[i][j] = A[i][j]/rs[i] ; Wb[i][j] = A[j][i]/cs[i]   (bf16 row-major)
__global__ void build_W(const float* __restrict__ A,
                        const float* __restrict__ rs, const float* __restrict__ cs,
                        unsigned short* __restrict__ Wf, unsigned short* __restrict__ Wb)
{
  const int total = NND * NND;
  for (int idx = blockIdx.x * blockDim.x + threadIdx.x; idx < total;
       idx += gridDim.x * blockDim.x) {
    const int i = idx / NND, j = idx % NND;
    Wf[idx] = f2bf(A[idx] / rs[i]);
    Wb[idx] = f2bf(A[(size_t)j * NND + i] / cs[i]);
  }
}

// Transpose gate weights to (64 x Kpad) bf16, zero-padding K >= Kreal
__global__ void wtrans(const float* __restrict__ W, int Kreal, int Kpad,
                       unsigned short* __restrict__ WT)
{
  const int total = HID * Kpad;
  for (int idx = blockIdx.x * blockDim.x + threadIdx.x; idx < total;
       idx += gridDim.x * blockDim.x) {
    const int o = idx / Kpad, k = idx % Kpad;
    WT[idx] = (k < Kreal) ? f2bf(W[(size_t)k * HID + o]) : (unsigned short)0;
  }
}

__global__ void zerof(float* __restrict__ p, int n) {
  for (int i = blockIdx.x * blockDim.x + threadIdx.x; i < n;
       i += gridDim.x * blockDim.x) p[i] = 0.0f;
}

// di[b,j] = x[b, T_in-1, j]
__global__ void copy_xlast(const float* __restrict__ x, float* __restrict__ di) {
  const int idx = blockIdx.x * blockDim.x + threadIdx.x;
  if (idx >= BB * NND) return;
  const int b = idx / NND, j = idx % NND;
  di[idx] = x[(size_t)b * (T_IN * NND) + (T_IN - 1) * NND + j];
}

// ---------------------------------------------------------------------------
extern "C" void kernel_launch(void* const* d_in, const int* in_sizes, int n_in,
                              void* d_out, int out_size, void* d_ws, size_t ws_size,
                              hipStream_t stream)
{
  (void)in_sizes; (void)n_in; (void)out_size; (void)ws_size;
  const float* x   = (const float*)d_in[0];
  const float* A   = (const float*)d_in[1];
  const float* Wr0 = (const float*)d_in[2];  const float* br0 = (const float*)d_in[3];
  const float* Wz0 = (const float*)d_in[4];  const float* bz0 = (const float*)d_in[5];
  const float* Wn0 = (const float*)d_in[6];  const float* bn0 = (const float*)d_in[7];
  const float* Wr1 = (const float*)d_in[8];  const float* br1 = (const float*)d_in[9];
  const float* Wz1 = (const float*)d_in[10]; const float* bz1 = (const float*)d_in[11];
  const float* Wn1 = (const float*)d_in[12]; const float* bn1 = (const float*)d_in[13];
  const float* Wp  = (const float*)d_in[14]; const float* bp  = (const float*)d_in[15];
  float* out = (float*)d_out;

  // ---- workspace carve-up (256B aligned) ----
  char* base = (char*)d_ws;
  size_t off = 0;
  auto take = [&](size_t bytes) -> void* {
    void* p = base + off;
    off = (off + bytes + 255) & ~(size_t)255;
    return p;
  };
  const int F0 = 1 + HID;            // 65
  const int F1 = 2 * HID;            // 128
  const int K0 = 352;                // 5*65=325 padded to mult of 32
  const int K1 = 5 * F1;             // 640
  const int C0 = BB * F0;            // 1040 (65 * 16)
  const int C1 = BB * F1;            // 2048

  unsigned short* Wf   = (unsigned short*)take((size_t)NND * NND * 2);
  unsigned short* Wb   = (unsigned short*)take((size_t)NND * NND * 2);
  unsigned short* WrT0 = (unsigned short*)take((size_t)HID * K0 * 2);
  unsigned short* WzT0 = (unsigned short*)take((size_t)HID * K0 * 2);
  unsigned short* WnT0 = (unsigned short*)take((size_t)HID * K0 * 2);
  unsigned short* WrT1 = (unsigned short*)take((size_t)HID * K1 * 2);
  unsigned short* WzT1 = (unsigned short*)take((size_t)HID * K1 * 2);
  unsigned short* WnT1 = (unsigned short*)take((size_t)HID * K1 * 2);
  float* rs = (float*)take(NND * 4);
  float* cs = (float*)take(NND * 4);
  float* h0 = (float*)take((size_t)NND * BB * HID * 4);
  float* h1 = (float*)take((size_t)NND * BB * HID * 4);
  unsigned short* X0T = (unsigned short*)take((size_t)C1 * NND * 2);
  unsigned short* X1T = (unsigned short*)take((size_t)C1 * NND * 2);
  unsigned short* G   = (unsigned short*)take((size_t)NND * BB * K1 * 2);
  float* Rb = (float*)take((size_t)NND * BB * HID * 4);
  float* Zb = (float*)take((size_t)NND * BB * HID * 4);
  float* Nb = (float*)take((size_t)NND * BB * HID * 4);
  float* di = (float*)take((size_t)BB * NND * 4);

  // ---- prep: normalized walk matrices + transposed bf16 gate weights ----
  rowcol_sums<<<NND, 256, 0, stream>>>(A, rs, cs);
  build_W<<<1024, 256, 0, stream>>>(A, rs, cs, Wf, Wb);
  wtrans<<<(HID * K0 + 255) / 256, 256, 0, stream>>>(Wr0, 325, K0, WrT0);
  wtrans<<<(HID * K0 + 255) / 256, 256, 0, stream>>>(Wz0, 325, K0, WzT0);
  wtrans<<<(HID * K0 + 255) / 256, 256, 0, stream>>>(Wn0, 325, K0, WnT0);
  wtrans<<<(HID * K1 + 255) / 256, 256, 0, stream>>>(Wr1, K1, K1, WrT1);
  wtrans<<<(HID * K1 + 255) / 256, 256, 0, stream>>>(Wz1, K1, K1, WzT1);
  wtrans<<<(HID * K1 + 255) / 256, 256, 0, stream>>>(Wn1, K1, K1, WnT1);
  zerof<<<2048, 256, 0, stream>>>(h0, NND * BB * HID);
  zerof<<<2048, 256, 0, stream>>>(h1, NND * BB * HID);

  // GEMM dispatch: NT=4 bulk blocks + NT=1 remainder 16-col tiles.
  auto gemm = [&](const unsigned short* Aop, int lda,
                  const unsigned short* Bt, int ldb,
                  int M, int Ncols, int K,
                  float* Cf, const float* bias, int act,
                  unsigned short* OutXT,
                  unsigned short* OutG, int g_pitch, int g_coloff, int Freal) {
    const int nfull = Ncols / 64;
    const int nrem  = (Ncols - nfull * 64) / 16;
    if (nfull > 0) {
      dim3 grid(M / 64, nfull);
      gemm_bf16_wmma<4><<<grid, 32, 0, stream>>>(
          Aop, lda, Bt, ldb, 0, K, Cf, HID, bias, act,
          OutXT, NND, OutG, g_pitch, g_coloff, Freal);
    }
    if (nrem > 0) {
      dim3 grid(M / 64, nrem);
      gemm_bf16_wmma<1><<<grid, 32, 0, stream>>>(
          Aop, lda, Bt, ldb, nfull * 64, K, Cf, HID, bias, act,
          OutXT, NND, OutG, g_pitch, g_coloff, Freal);
    }
  };

  // One DCGRU cell. layer 0: xl scalar input (FA=1); layer 1: xl = h0 (FA=64).
  auto cell = [&](int layer, const float* xlp, int xstride) {
    const int F  = layer ? F1 : F0;
    const int Kg = layer ? K1 : K0;
    const int Cc = layer ? C1 : C0;
    const int FA = layer ? HID : 1;
    const int pad = Kg - 5 * F;
    const float* hB = layer ? h1 : h0;
    float* hUpd = layer ? h1 : h0;
    const unsigned short* WrT = layer ? WrT1 : WrT0;
    const unsigned short* WzT = layer ? WzT1 : WzT0;
    const unsigned short* WnT = layer ? WnT1 : WnT0;
    const float* br = layer ? br1 : br0;
    const float* bz = layer ? bz1 : bz0;
    const float* bn = layer ? bn1 : bn0;
    const int cgrid = (NND * BB * (F + pad) + 255) / 256;

    // pass 1: xh = [xl | h]; diffusion; r and z gates
    concat_kernel<<<cgrid, 256, 0, stream>>>(xlp, xstride, FA, hB, nullptr,
                                             F, pad, Kg, X0T, G);
    gemm(Wf, NND, X0T, NND, NND, Cc, NND, nullptr, nullptr, 0, X1T, G, Kg, 1 * F, F);
    gemm(Wf, NND, X1T, NND, NND, Cc, NND, nullptr, nullptr, 0, nullptr, G, Kg, 2 * F, F);
    gemm(Wb, NND, X0T, NND, NND, Cc, NND, nullptr, nullptr, 0, X1T, G, Kg, 3 * F, F);
    gemm(Wb, NND, X1T, NND, NND, Cc, NND, nullptr, nullptr, 0, nullptr, G, Kg, 4 * F, F);
    gemm(G, Kg, WrT, Kg, NND * BB, HID, Kg, Rb, br, 1, nullptr, nullptr, 0, 0, 1);
    gemm(G, Kg, WzT, Kg, NND * BB, HID, Kg, Zb, bz, 1, nullptr, nullptr, 0, 0, 1);

    // pass 2: xrh = [xl | r*h]; diffusion; n gate; state update
    concat_kernel<<<cgrid, 256, 0, stream>>>(xlp, xstride, FA, hB, Rb,
                                             F, pad, Kg, X0T, G);
    gemm(Wf, NND, X0T, NND, NND, Cc, NND, nullptr, nullptr, 0, X1T, G, Kg, 1 * F, F);
    gemm(Wf, NND, X1T, NND, NND, Cc, NND, nullptr, nullptr, 0, nullptr, G, Kg, 2 * F, F);
    gemm(Wb, NND, X0T, NND, NND, Cc, NND, nullptr, nullptr, 0, X1T, G, Kg, 3 * F, F);
    gemm(Wb, NND, X1T, NND, NND, Cc, NND, nullptr, nullptr, 0, nullptr, G, Kg, 4 * F, F);
    gemm(G, Kg, WnT, Kg, NND * BB, HID, Kg, Nb, bn, 2, nullptr, nullptr, 0, 0, 1);
    gru_update<<<2048, 256, 0, stream>>>(hUpd, Zb, Nb, NND * BB * HID);
  };

  // ---- encoder ----
  for (int t = 0; t < T_IN; ++t) {
    cell(0, x + (size_t)t * NND, T_IN * NND);
    cell(1, h0, 0);
  }
  // ---- decoder ----
  copy_xlast<<<(BB * NND + 255) / 256, 256, 0, stream>>>(x, di);
  for (int t = 0; t < T_OUT; ++t) {
    cell(0, di, NND);
    cell(1, h0, 0);
    head_kernel<<<(BB * NND + 255) / 256, 256, 0, stream>>>(
        h1, Wp, bp, out + (size_t)t * NND, di);
  }
}
